// MSDeformAttnTransformerEncoderLayer_88630945120262
// MI455X (gfx1250) — matrix operand order repaired and checked by reading, not compiled, not run on hardware
//
#include <hip/hip_runtime.h>
#include <math.h>

// ---------------------------------------------------------------------------
// Problem constants (from the reference)
// ---------------------------------------------------------------------------
constexpr int BQ      = 2;
constexpr int LQTOT   = 21760;          // 128^2 + 64^2 + 32^2 + 16^2
constexpr int BLQ     = BQ * LQTOT;     // 43520 token rows
constexpr int DM      = 256;            // d_model
constexpr int DF      = 1024;           // d_ffn
constexpr int NHEADS  = 8;
constexpr int NLVL    = 4;
constexpr int NPTS    = 4;
constexpr int HD      = 32;             // head dim == wave32 width

constexpr size_t EL   = (size_t)BLQ * DM;    // 11,141,120
constexpr size_t ELA  = (size_t)BLQ * (NHEADS * NLVL * NPTS); // attn logits (BLQ*128)
constexpr size_t ELF  = (size_t)BLQ * DF;

// ---------------------------------------------------------------------------
// Types
// ---------------------------------------------------------------------------
typedef __attribute__((ext_vector_type(16))) __bf16          v16bf;
typedef __attribute__((ext_vector_type(8)))  float           v8f;
typedef __attribute__((ext_vector_type(8)))  unsigned short  ushort8;

union FragBF {            // 32 bytes: two 16B global loads -> one WMMA operand
    ushort8 u[2];
    v16bf   v;
};

__device__ __forceinline__ unsigned short f32_to_bf16(float f) {
    unsigned int u = __float_as_uint(f);
    unsigned int lsb = (u >> 16) & 1u;
    u += 0x7fffu + lsb;                  // round to nearest even
    return (unsigned short)(u >> 16);
}

// ---------------------------------------------------------------------------
// Elementwise prep: query = src + pos ; bf16 copies of src and query
// ---------------------------------------------------------------------------
__global__ __launch_bounds__(256) void prep_act_kernel(
    const float* __restrict__ src, const float* __restrict__ pos,
    unsigned short* __restrict__ src_b, unsigned short* __restrict__ q_b,
    size_t n)
{
    size_t i = (size_t)blockIdx.x * blockDim.x + threadIdx.x;
    if (i >= n) return;
    float s = src[i];
    float q = s + pos[i];
    src_b[i] = f32_to_bf16(s);
    q_b[i]   = f32_to_bf16(q);
}

// Weight convert + transpose:  W[K,N] fp32  ->  Wt[N,K] bf16
__global__ __launch_bounds__(256) void convw_kernel(
    const float* __restrict__ W, unsigned short* __restrict__ Wt, int K, int N)
{
    int i = blockIdx.x * blockDim.x + threadIdx.x;
    if (i >= K * N) return;
    int k = i / N, n = i - k * N;
    Wt[(size_t)n * K + k] = f32_to_bf16(W[i]);
}

// ---------------------------------------------------------------------------
// WMMA bf16 GEMM:  C[M,N] = A[M,K] @ Wt[N,K]^T + bias  (optional ReLU)
// Block = 256 threads = 8 waves (4 x 2), block tile 128 x 128,
// wave tile 32 x 64 (2 x 4 WMMA tiles), K step 32.  L2-resident operands,
// fragments loaded directly from global in ISA wave32 layout.
// ---------------------------------------------------------------------------
constexpr int GBM = 128;
constexpr int GBN = 128;

__global__ __launch_bounds__(256) void gemm_bf16_kernel(
    const unsigned short* __restrict__ A,    // [M,K] bf16 bits, row-major
    const unsigned short* __restrict__ Wt,   // [N,K] bf16 bits, row-major
    const float* __restrict__ bias,          // [N] or null
    float* __restrict__ Cf,                  // [M,N] fp32 or null
    unsigned short* __restrict__ Cb,         // [M,N] bf16 or null
    int M, int N, int K, int relu)
{
    const int lane = threadIdx.x & 31;
    const int wave = threadIdx.x >> 5;
    const int wm   = wave >> 1;              // 0..3
    const int wn   = wave & 1;               // 0..1
    const int m0   = blockIdx.y * GBM + wm * 32;
    const int n0   = blockIdx.x * GBN + wn * 64;
    const int hl   = lane >> 4;              // which half of the wave
    const int l16  = lane & 15;

    v8f acc[2][4];
    #pragma unroll
    for (int i = 0; i < 2; ++i)
        #pragma unroll
        for (int j = 0; j < 4; ++j)
            #pragma unroll
            for (int r = 0; r < 8; ++r) acc[i][j][r] = 0.0f;

    for (int kk = 0; kk < K; kk += 32) {
        FragBF a[2], b[4];
        // A fragment: lane l16 = row, elems 0..7 -> K = kk+hl*8+e,
        //             elems 8..15 -> K = kk+16+hl*8+(e-8)   (ISA 7.12.2)
        #pragma unroll
        for (int mt = 0; mt < 2; ++mt) {
            int row = m0 + mt * 16 + l16; if (row >= M) row = M - 1;
            const unsigned short* p = A + (size_t)row * K + kk + hl * 8;
            a[mt].u[0] = *(const ushort8*)(p);
            a[mt].u[1] = *(const ushort8*)(p + 16);
        }
        // B fragment: lane l16 = column, same K layout as A
        #pragma unroll
        for (int nt = 0; nt < 4; ++nt) {
            int col = n0 + nt * 16 + l16; if (col >= N) col = N - 1;
            const unsigned short* p = Wt + (size_t)col * K + kk + hl * 8;
            b[nt].u[0] = *(const ushort8*)(p);
            b[nt].u[1] = *(const ushort8*)(p + 16);
        }
        #pragma unroll
        for (int mt = 0; mt < 2; ++mt)
            #pragma unroll
            for (int nt = 0; nt < 4; ++nt)
                acc[mt][nt] = __builtin_amdgcn_wmma_f32_16x16x32_bf16(
                    false, a[mt].v, false, b[nt].v,
                    (short)0, acc[mt][nt], false, false);
    }

    // Epilogue: C/D layout -> VGPR r: M = r + hl*8, N = l16
    #pragma unroll
    for (int mt = 0; mt < 2; ++mt)
        #pragma unroll
        for (int nt = 0; nt < 4; ++nt) {
            int col = n0 + nt * 16 + l16;
            if (col >= N) continue;
            float bv = bias ? bias[col] : 0.0f;
            #pragma unroll
            for (int r = 0; r < 8; ++r) {
                int row = m0 + mt * 16 + hl * 8 + r;
                if (row >= M) continue;
                float v = acc[mt][nt][r] + bv;
                if (relu) v = fmaxf(v, 0.0f);
                size_t idx = (size_t)row * N + col;
                if (Cf) Cf[idx] = v;
                if (Cb) Cb[idx] = f32_to_bf16(v);
            }
        }
}

// ---------------------------------------------------------------------------
// Multi-scale deformable attention sampling.
// One wave per (b, q, head); lane = channel (HEAD_DIM == 32).
// ---------------------------------------------------------------------------
__global__ __launch_bounds__(256) void msda_kernel(
    const float* __restrict__ value,     // [B*LQ, 256]  (b, loc, h*32+d)
    const float* __restrict__ off,       // [B*LQ, 256]  (h, l, p, 2)
    const float* __restrict__ attn,      // [B*LQ, 128]  (h, l*4+p)
    const float* __restrict__ refp,      // [B*LQ, 4, 2]
    const int*   __restrict__ ss,        // [4,2] (H, W)
    const int*   __restrict__ lsi,       // [4]
    unsigned short* __restrict__ out_b)  // [B*LQ, 256] bf16 (b, q, h*32+d)
{
    int gid  = blockIdx.x * (blockDim.x >> 5) + (threadIdx.x >> 5);
    int lane = threadIdx.x & 31;
    if (gid >= BLQ * NHEADS) return;
    int bq = gid >> 3;                    // b*LQ + q
    int h  = gid & 7;
    int b  = bq / LQTOT;

    // softmax over the 16 (level,point) logits for this head
    const float* ap = attn + (size_t)bq * (NHEADS * 16) + h * 16;
    float mx = -3.0e38f;
    #pragma unroll
    for (int i = 0; i < 16; ++i) mx = fmaxf(mx, ap[i]);
    float w[16], s = 0.0f;
    #pragma unroll
    for (int i = 0; i < 16; ++i) { w[i] = __expf(ap[i] - mx); s += w[i]; }
    float inv = 1.0f / s;

    const float* op = off + (size_t)bq * DM + h * (NLVL * NPTS * 2);
    float acc = 0.0f;

    for (int l = 0; l < NLVL; ++l) {
        int Hh = ss[2 * l], Ww = ss[2 * l + 1];
        int st = lsi[l];
        float rx = refp[((size_t)bq * NLVL + l) * 2 + 0];
        float ry = refp[((size_t)bq * NLVL + l) * 2 + 1];
        const float* vbase = value + ((size_t)(b * LQTOT + st)) * DM + h * HD + lane;
        #pragma unroll
        for (int p = 0; p < NPTS; ++p) {
            float ox = op[(l * NPTS + p) * 2 + 0];
            float oy = op[(l * NPTS + p) * 2 + 1];
            // loc = ref + off/(W,H);  x = loc_x*W - 0.5
            float x = rx * (float)Ww + ox - 0.5f;
            float y = ry * (float)Hh + oy - 0.5f;
            float aw = w[l * NPTS + p] * inv;

            float x0f = floorf(x), y0f = floorf(y);
            int   x0 = (int)x0f,   y0 = (int)y0f;
            float wx1 = x - x0f, wy1 = y - y0f;
            float wx0 = 1.0f - wx1, wy0 = 1.0f - wy1;

            #pragma unroll
            for (int c = 0; c < 4; ++c) {
                int   xi = x0 + (c & 1);
                int   yi = y0 + (c >> 1);
                float wt = ((c & 1) ? wx1 : wx0) * ((c >> 1) ? wy1 : wy0);
                if (xi >= 0 && xi < Ww && yi >= 0 && yi < Hh)
                    acc += aw * wt * vbase[((size_t)yi * Ww + xi) * DM];
            }
        }
    }
    out_b[(size_t)bq * DM + h * HD + lane] = f32_to_bf16(acc);
}

// ---------------------------------------------------------------------------
// Fused residual + LayerNorm: out = LN(a + c) * g + b ; one wave per row(256)
// ---------------------------------------------------------------------------
__global__ __launch_bounds__(256) void add_ln_kernel(
    const float* __restrict__ a, const float* __restrict__ c,
    const float* __restrict__ g, const float* __restrict__ bta,
    float* __restrict__ outF, unsigned short* __restrict__ outB)
{
    int row  = blockIdx.x * (blockDim.x >> 5) + (threadIdx.x >> 5);
    int lane = threadIdx.x & 31;
    if (row >= BLQ) return;
    size_t base = (size_t)row * DM;

    float v[8];
    float s = 0.0f;
    #pragma unroll
    for (int j = 0; j < 8; ++j) {
        int col = j * 32 + lane;
        v[j] = a[base + col] + c[base + col];
        s += v[j];
    }
    #pragma unroll
    for (int o = 16; o > 0; o >>= 1) s += __shfl_xor(s, o, 32);
    float mu = s * (1.0f / 256.0f);

    float var = 0.0f;
    #pragma unroll
    for (int j = 0; j < 8; ++j) { float d = v[j] - mu; var += d * d; }
    #pragma unroll
    for (int o = 16; o > 0; o >>= 1) var += __shfl_xor(var, o, 32);
    float r = rsqrtf(var * (1.0f / 256.0f) + 1e-5f);

    #pragma unroll
    for (int j = 0; j < 8; ++j) {
        int col = j * 32 + lane;
        float o = (v[j] - mu) * r * g[col] + bta[col];
        outF[base + col] = o;
        if (outB) outB[base + col] = f32_to_bf16(o);
    }
}

// ---------------------------------------------------------------------------
// Host-side orchestration
// ---------------------------------------------------------------------------
static inline dim3 gemm_grid(int M, int N) {
    return dim3((N + GBN - 1) / GBN, (M + GBM - 1) / GBM);
}

extern "C" void kernel_launch(void* const* d_in, const int* in_sizes, int n_in,
                              void* d_out, int out_size, void* d_ws, size_t ws_size,
                              hipStream_t stream) {
    (void)in_sizes; (void)n_in; (void)out_size; (void)ws_size;

    const float* src   = (const float*)d_in[0];
    const float* pos   = (const float*)d_in[1];
    const float* refp  = (const float*)d_in[2];
    const float* W_off = (const float*)d_in[3];
    const float* b_off = (const float*)d_in[4];
    const float* W_att = (const float*)d_in[5];
    const float* b_att = (const float*)d_in[6];
    const float* W_val = (const float*)d_in[7];
    const float* b_val = (const float*)d_in[8];
    const float* W_out = (const float*)d_in[9];
    const float* b_out = (const float*)d_in[10];
    const float* ln1g  = (const float*)d_in[11];
    const float* ln1b  = (const float*)d_in[12];
    const float* W1    = (const float*)d_in[13];
    const float* b1    = (const float*)d_in[14];
    const float* W2    = (const float*)d_in[15];
    const float* b2    = (const float*)d_in[16];
    const float* ln2g  = (const float*)d_in[17];
    const float* ln2b  = (const float*)d_in[18];
    const int*   ss    = (const int*)d_in[19];
    const int*   lsi   = (const int*)d_in[20];
    float*       out   = (float*)d_out;

    char* ws = (char*)d_ws;
    // workspace layout (byte offsets); buffers reused once their producer dies
    size_t o_val  = 0;                    // EL*4   value_f32  -> later ffn2_f32
    size_t o_off  = o_val  + EL * 4;      // EL*4   off_f32    -> later h_f32
    size_t o_attn = o_off  + EL * 4;      // ELA*4  attn_f32   -> later h_bf16 (EL*2, same bytes)
    size_t o_srcb = o_attn + EL * 2;      // EL*2   src_bf16   -> later msda_bf16
    size_t o_qb   = o_srcb + EL * 2;      // EL*2   query_bf16
    size_t o_relu = o_qb   + EL * 2;      // ELF*2  relu_bf16
    size_t o_aout = o_relu + ELF * 2;     // EL*4   attn-out fp32
    size_t o_w    = o_aout + EL * 4;      // bf16 transposed weights

    unsigned short* wt_val = (unsigned short*)(ws + o_w);
    unsigned short* wt_off = wt_val + (size_t)DM * DM;
    unsigned short* wt_att = wt_off + (size_t)DM * DM;
    unsigned short* wt_out = wt_att + (size_t)DM * 128;
    unsigned short* wt_1   = wt_out + (size_t)DM * DM;
    unsigned short* wt_2   = wt_1   + (size_t)DM * DF;

    float*          value_f = (float*)(ws + o_val);
    float*          off_f   = (float*)(ws + o_off);
    float*          attn_f  = (float*)(ws + o_attn);
    unsigned short* src_b   = (unsigned short*)(ws + o_srcb);
    unsigned short* q_b     = (unsigned short*)(ws + o_qb);
    unsigned short* msda_b  = (unsigned short*)(ws + o_srcb);  // reuse src_b
    float*          aout_f  = (float*)(ws + o_aout);
    float*          h_f     = (float*)(ws + o_off);            // reuse off_f
    unsigned short* h_b     = (unsigned short*)(ws + o_attn);  // reuse attn_f
    unsigned short* relu_b  = (unsigned short*)(ws + o_relu);
    float*          ffn2_f  = (float*)(ws + o_val);            // reuse value_f

    // 1) query = src + pos; bf16 activations
    prep_act_kernel<<<(unsigned)((EL + 255) / 256), 256, 0, stream>>>(
        src, pos, src_b, q_b, EL);

    // 2) weights -> bf16 [N,K]
    convw_kernel<<<(DM * DM  + 255) / 256, 256, 0, stream>>>(W_val, wt_val, DM, DM);
    convw_kernel<<<(DM * DM  + 255) / 256, 256, 0, stream>>>(W_off, wt_off, DM, DM);
    convw_kernel<<<(DM * 128 + 255) / 256, 256, 0, stream>>>(W_att, wt_att, DM, 128);
    convw_kernel<<<(DM * DM  + 255) / 256, 256, 0, stream>>>(W_out, wt_out, DM, DM);
    convw_kernel<<<(DM * DF  + 255) / 256, 256, 0, stream>>>(W1,    wt_1,   DM, DF);
    convw_kernel<<<(DF * DM  + 255) / 256, 256, 0, stream>>>(W2,    wt_2,   DF, DM);

    // 3) projections (WMMA bf16, fp32 accumulate)
    gemm_bf16_kernel<<<gemm_grid(BLQ, DM), 256, 0, stream>>>(
        src_b, wt_val, b_val, value_f, (unsigned short*)nullptr, BLQ, DM, DM, 0);
    gemm_bf16_kernel<<<gemm_grid(BLQ, DM), 256, 0, stream>>>(
        q_b, wt_off, b_off, off_f, (unsigned short*)nullptr, BLQ, DM, DM, 0);
    gemm_bf16_kernel<<<gemm_grid(BLQ, 128), 256, 0, stream>>>(
        q_b, wt_att, b_att, attn_f, (unsigned short*)nullptr, BLQ, 128, DM, 0);

    // 4) deformable sampling (wave per (b,q,head), lane = channel)
    msda_kernel<<<BLQ, 256, 0, stream>>>(value_f, off_f, attn_f, refp, ss, lsi, msda_b);

    // 5) output projection -> residual+LN1 -> FFN -> residual+LN2
    gemm_bf16_kernel<<<gemm_grid(BLQ, DM), 256, 0, stream>>>(
        msda_b, wt_out, b_out, aout_f, (unsigned short*)nullptr, BLQ, DM, DM, 0);
    add_ln_kernel<<<(BLQ + 7) / 8, 256, 0, stream>>>(src, aout_f, ln1g, ln1b, h_f, h_b);
    gemm_bf16_kernel<<<gemm_grid(BLQ, DF), 256, 0, stream>>>(
        h_b, wt_1, b1, (float*)nullptr, relu_b, BLQ, DF, DM, 1);
    gemm_bf16_kernel<<<gemm_grid(BLQ, DM), 256, 0, stream>>>(
        relu_b, wt_2, b2, ffn2_f, (unsigned short*)nullptr, BLQ, DM, DF, 0);
    add_ln_kernel<<<(BLQ + 7) / 8, 256, 0, stream>>>(h_f, ffn2_f, ln2g, ln2b, out,
                                                     (unsigned short*)nullptr);
}
